// NodeBlock_36069135352226
// MI455X (gfx1250) — matrix operand order repaired and checked by reading, not compiled, not run on hardware
//
#include <hip/hip_runtime.h>
#include <hip/hip_bf16.h>

typedef _Float16 v16h __attribute__((ext_vector_type(16)));
typedef _Float16 v8h  __attribute__((ext_vector_type(8)));
typedef float    v8f  __attribute__((ext_vector_type(8)));

#define D_NODE 64
#define D_IN   128
#define D_OUT  128
#define TILE_E 16
#define ITERS  8
#define EDGES_PER_BLOCK (TILE_E * ITERS)
#define LN_EPS 1e-5f

// ---------------- zero the aggregation buffer ----------------
__global__ __launch_bounds__(256) void nb_zero_kernel(float* __restrict__ p, long long n) {
  long long idx = (long long)blockIdx.x * blockDim.x + threadIdx.x;
  if (idx < n) p[idx] = 0.f;
}

// ---------------- edge kernel: gather+concat -> WMMA GEMM -> LN -> gate -> scatter ----------------
__global__ __launch_bounds__(256) void nb_edge_kernel(
    const float* __restrict__ node_emb, const float* __restrict__ edge_emb,
    const int*   __restrict__ idx,      const float* __restrict__ W,
    const float* __restrict__ bias,     const float* __restrict__ gamma1,
    const float* __restrict__ beta1,    float* __restrict__ agg, long long E) {
  __shared__ __align__(32) _Float16 sW[D_OUT * D_IN];   // W[o][k], f16, K contiguous
  __shared__ __align__(32) _Float16 sA[TILE_E * D_IN];  // concat input tile, f16
  __shared__ float sY[TILE_E * D_OUT];                  // GEMM output tile, f32

  const int t     = threadIdx.x;   // 0..255 (8 wave32)
  const int wv    = t >> 5;
  const int lane  = t & 31;
  const int lhalf = lane >> 4;     // 0 or 1
  const int l15   = lane & 15;

  // y[e][o] = sum_k c1[e][k] * W[o][k]  ->  B[k][n] = W[n][k]; keep W row-major so each
  // lane's 16-K run at fixed n is one contiguous 32B v16h LDS load.
  for (int u = t; u < D_OUT * D_IN; u += 256) sW[u] = (_Float16)W[u];

  const long long ebase = (long long)blockIdx.x * EDGES_PER_BLOCK;
  const int   ncol = wv * 16 + l15;   // this wave's output column for B / C / D
  const float bv   = bias[ncol];

  for (int it = 0; it < ITERS; ++it) {
    const long long e0 = ebase + (long long)it * TILE_E;
    __syncthreads();  // sW ready (1st iter) / previous iter fully consumed

    // stage 16 rows of concat(node[i[e]], edge[e]) as f16
    for (int u = t; u < TILE_E * D_IN; u += 256) {
      const int er = u >> 7;
      const int f  = u & 127;
      const long long e = e0 + er;
      float v = 0.f;
      if (e < E) {
        if (f < D_NODE) v = node_emb[(long long)idx[e] * D_NODE + f];
        else            v = edge_emb[e * D_NODE + (f - D_NODE)];
      }
      sA[u] = (_Float16)v;
    }
    __syncthreads();

    // one 16x16 output tile per wave; K=128 via 4x v_wmma_f32_16x16x32_f16
    v8f acc = {};
#pragma unroll
    for (int kc = 0; kc < 4; ++kc) {
      const int kb = kc * 32;
      // A layout (contiguous K, matches B): lane half 0 -> K = kb+0..15
      //                                     lane half 1 -> K = kb+16..31
      const v8h lo = *(const v8h*)&sA[l15 * D_IN + kb + lhalf * 16];
      const v8h hi = *(const v8h*)&sA[l15 * D_IN + kb + lhalf * 16 + 8];
      const v16h afrag = __builtin_shufflevector(lo, hi,
          0, 1, 2, 3, 4, 5, 6, 7, 8, 9, 10, 11, 12, 13, 14, 15);
      // B layout: lanes 0-15 hold K=kb+0..15, lanes 16-31 hold K=kb+16..31, N = lane&15
      const v16h bfrag = *(const v16h*)&sW[ncol * D_IN + kb + lhalf * 16];
      acc = __builtin_amdgcn_wmma_f32_16x16x32_f16(
          false, afrag, false, bfrag, (short)0, acc, false, false);
    }

    // C/D layout: VGPR r -> M = r + 8*lhalf, N = lane&15; add bias, stage f32
#pragma unroll
    for (int r = 0; r < 8; ++r) {
      const int mrow = r + lhalf * 8;
      sY[mrow * D_OUT + ncol] = acc[r] + bv;
    }
    __syncthreads();

    // LayerNorm(128) per edge row: 16 threads per row, 8 features each.
    const int row = t >> 4;
    const int c   = t & 15;
    const long long e = e0 + row;
    float yv[8];
    float s = 0.f;
#pragma unroll
    for (int j = 0; j < 8; ++j) { yv[j] = sY[row * D_OUT + c + 16 * j]; s += yv[j]; }
#pragma unroll
    for (int m = 8; m >= 1; m >>= 1) s += __shfl_xor(s, m, 32);  // stays in 16-lane group
    const float mu = s * (1.f / 128.f);
    float v2 = 0.f;
#pragma unroll
    for (int j = 0; j < 8; ++j) { const float d = yv[j] - mu; v2 += d * d; }
#pragma unroll
    for (int m = 8; m >= 1; m >>= 1) v2 += __shfl_xor(v2, m, 32);
    const float rstd = rsqrtf(v2 * (1.f / 128.f) + LN_EPS);
    float yn[8];
#pragma unroll
    for (int j = 0; j < 8; ++j) {
      const int f = c + 16 * j;
      yn[j] = (yv[j] - mu) * rstd * gamma1[f] + beta1[f];
    }
    // gate: msg[m] = sigmoid(yn[m]) * tanh(yn[m+64]); m = c+16j pairs with c+16(j+4)
    if (e < E) {
      const long long nd = idx[e];
#pragma unroll
      for (int j = 0; j < 4; ++j) {
        const int m = c + 16 * j;
        const float msg = (1.f / (1.f + __expf(-yn[j]))) * tanhf(yn[j + 4]);
        atomicAdd(&agg[nd * D_NODE + m], msg);  // global_atomic_add_f32, L2-resident
      }
    }
  }
}

// ---------------- node kernel: LN(64) + residual tanh ----------------
__global__ __launch_bounds__(256) void nb_node_kernel(
    const float* __restrict__ node_emb, const float* __restrict__ agg,
    const float* __restrict__ gamma2,   const float* __restrict__ beta2,
    float* __restrict__ out, long long N) {
  const int lane = threadIdx.x & 31;
  const long long n = (long long)blockIdx.x * 8 + (threadIdx.x >> 5);  // one wave32 per node
  if (n >= N) return;
  const float a0 = agg[n * D_NODE + lane];
  const float a1 = agg[n * D_NODE + lane + 32];
  float s = a0 + a1;
#pragma unroll
  for (int m = 16; m >= 1; m >>= 1) s += __shfl_xor(s, m, 32);
  const float mu = s * (1.f / 64.f);
  const float d0 = a0 - mu, d1 = a1 - mu;
  float v2 = d0 * d0 + d1 * d1;
#pragma unroll
  for (int m = 16; m >= 1; m >>= 1) v2 += __shfl_xor(v2, m, 32);
  const float rstd = rsqrtf(v2 * (1.f / 64.f) + LN_EPS);
  out[n * D_NODE + lane] =
      tanhf(node_emb[n * D_NODE + lane] + d0 * rstd * gamma2[lane] + beta2[lane]);
  out[n * D_NODE + lane + 32] =
      tanhf(node_emb[n * D_NODE + lane + 32] + d1 * rstd * gamma2[lane + 32] + beta2[lane + 32]);
}

// ---------------- launcher ----------------
extern "C" void kernel_launch(void* const* d_in, const int* in_sizes, int n_in,
                              void* d_out, int out_size, void* d_ws, size_t ws_size,
                              hipStream_t stream) {
  const float* node_emb = (const float*)d_in[0];
  const float* edge_emb = (const float*)d_in[1];
  const float* W        = (const float*)d_in[2];
  const float* b        = (const float*)d_in[3];
  const float* gamma1   = (const float*)d_in[4];
  const float* beta1    = (const float*)d_in[5];
  const float* gamma2   = (const float*)d_in[6];
  const float* beta2    = (const float*)d_in[7];
  const int*   idx      = (const int*)d_in[8];

  const long long N = (long long)in_sizes[0] / D_NODE;
  const long long E = (long long)in_sizes[1] / D_NODE;

  float* agg = (float*)d_ws;         // N*64 f32 = 25.6 MB scratch
  float* out = (float*)d_out;

  const long long aggN = N * D_NODE;
  nb_zero_kernel<<<(unsigned)((aggN + 255) / 256), 256, 0, stream>>>(agg, aggN);

  const unsigned eblocks = (unsigned)((E + EDGES_PER_BLOCK - 1) / EDGES_PER_BLOCK);
  nb_edge_kernel<<<eblocks, 256, 0, stream>>>(node_emb, edge_emb, idx, W, b,
                                              gamma1, beta1, agg, E);

  const unsigned nblocks = (unsigned)((N + 7) / 8);
  nb_node_kernel<<<nblocks, 256, 0, stream>>>(node_emb, agg, gamma2, beta2, out, N);
}
